// LIFEAttention_67585605370563
// MI455X (gfx1250) — compile-verified, matrix operand order checked by reading, not run on hardware
//
#include <hip/hip_runtime.h>
#include <math.h>
#include <stdint.h>

typedef _Float16 h16;
typedef __attribute__((ext_vector_type(16))) _Float16 v16h;
typedef __attribute__((ext_vector_type(8)))  float    v8f;

__device__ __forceinline__ v8f wmma_f16(v16h a, v16h b, v8f c) {
  // D = A(16x32 f16) * B(32x16 f16) + C(16x16 f32)
  return __builtin_amdgcn_wmma_f32_16x16x32_f16(false, a, false, b, (short)0, c, false, false);
}

// K-offset of fragment element e for half-select hs (ISA 16-bit A/B layout)
__device__ __forceinline__ int frag_k(int e, int hs) {
  return ((e >> 3) << 4) + (hs << 3) + (e & 7);
}

// Async 16-byte copy global -> LDS (GVS mode: saddr base + 32-bit vgpr offset).
// Per-lane: LDS[lds_addr] = MEM[base + voff] (16B). Tracked by ASYNCcnt.
__device__ __forceinline__ void async_cp16(uint32_t lds_addr, const void* base, uint32_t voff) {
  asm volatile("global_load_async_to_lds_b128 %0, %1, %2"
               :: "v"(lds_addr), "v"(voff), "s"(base) : "memory");
}
__device__ __forceinline__ void async_wait0() {
  asm volatile("s_wait_asynccnt 0x0" ::: "memory");
}
__device__ __forceinline__ uint32_t lds_off(const void* p) {
  return (uint32_t)(uintptr_t)p;  // low 32 bits of generic LDS ptr = wave LDS addr
}

// ---------------------------------------------------------------------------
// Depthwise k x k conv (SAME, zero pad) on (B,C,24,24); x is (B,N,C) fp32.
// Writes token-major fp16: y[t*C + c].
// ---------------------------------------------------------------------------
__global__ void dwconv_kernel(const float* __restrict__ x, const float* __restrict__ dw,
                              h16* __restrict__ y, int ks, int total) {
  for (int idx = blockIdx.x * blockDim.x + threadIdx.x; idx < total;
       idx += gridDim.x * blockDim.x) {
    const int c = idx % 768;
    const int t = idx / 768;
    const int b = t / 576, n = t % 576;
    const int h = n / 24, w = n % 24;
    const int r = ks >> 1;
    float acc = 0.f;
    for (int ky = 0; ky < ks; ++ky) {
      const int hh = h + ky - r;
      if ((unsigned)hh >= 24u) continue;
      for (int kx = 0; kx < ks; ++kx) {
        const int ww = w + kx - r;
        if ((unsigned)ww >= 24u) continue;
        acc += x[((size_t)b * 576 + hh * 24 + ww) * 768 + c] *
               dw[c * ks * ks + ky * ks + kx];
      }
    }
    y[idx] = (h16)acc;
  }
}

// ---------------------------------------------------------------------------
// Weight convert+transpose: W (N rows x K cols fp32, row-major) -> Bh[k*N+n] fp16
// ---------------------------------------------------------------------------
__global__ void wcvt_kernel(const float* __restrict__ W, h16* __restrict__ Bh,
                            int Nn, int Kk, int total) {
  for (int idx = blockIdx.x * blockDim.x + threadIdx.x; idx < total;
       idx += gridDim.x * blockDim.x) {
    const int k = idx / Nn;
    const int n = idx % Nn;
    Bh[idx] = (h16)W[(size_t)n * Kk + k];
  }
}

// ---------------------------------------------------------------------------
// WMMA GEMM: D = A(MxK f16, row-major) * Bm(KxN f16, [k][n]) (+ epilogue)
// 64x64 tile per block, 4 waves, each wave 32x32 (2x2 WMMA frags), K-step 32.
// Tiles staged via GLOBAL_LOAD_ASYNC_TO_LDS_B128 (ASYNCcnt).
// EPI 0: BN(inference)+exact GELU, fp16 write at column offset (branch concat)
// EPI 1: +bias0(+bias1), scatter into (B, head, 576, 64) fp16 head layout
// EPI 2: +bias0, fp32 write (final output)
// ---------------------------------------------------------------------------
template <int EPI>
__launch_bounds__(128)
__global__ void gemm_wmma_kernel(const h16* __restrict__ A, const h16* __restrict__ Bm,
                                 int M, int N, int K,
                                 h16* __restrict__ Dh, float* __restrict__ Df,
                                 int ldD, int colOff,
                                 const float* __restrict__ bn_g, const float* __restrict__ bn_b,
                                 const float* __restrict__ bn_m, const float* __restrict__ bn_v,
                                 const float* __restrict__ bias0, const float* __restrict__ bias1) {
  __shared__ h16 As[64 * 40];  // 64 rows x 32 k (+8 pad)
  __shared__ h16 Bs[32 * 72];  // 32 k x 64 n (+8 pad)

  const int tid = threadIdx.x;
  const int wv = tid >> 5, lane = tid & 31;
  const int hs = lane >> 4, ln16 = lane & 15;
  const int blkM = blockIdx.x * 64, blkN = blockIdx.y * 64;
  const int wm = (wv >> 1) * 32, wn = (wv & 1) * 32;

  v8f vz = {};
  v8f acc[2][2];
#pragma unroll
  for (int mt = 0; mt < 2; ++mt)
#pragma unroll
    for (int nt = 0; nt < 2; ++nt) acc[mt][nt] = vz;

  const int rA = tid >> 1, kbA = (tid & 1) * 16;  // A: 64 rows x 32, 16 halfs/thread
  const int rB = tid >> 2, nbB = (tid & 3) * 16;  // B: 32 rows x 64, 16 halfs/thread
  const uint32_t ldsA = lds_off(As + rA * 40 + kbA);
  const uint32_t ldsB = lds_off(Bs + rB * 72 + nbB);

  for (int k0 = 0; k0 < K; k0 += 32) {
    const uint32_t voffA = (uint32_t)((((size_t)(blkM + rA)) * K + k0 + kbA) * sizeof(h16));
    const uint32_t voffB = (uint32_t)((((size_t)(k0 + rB)) * N + blkN + nbB) * sizeof(h16));
    async_cp16(ldsA,      A,  voffA);
    async_cp16(ldsA + 16, A,  voffA + 16);
    async_cp16(ldsB,      Bm, voffB);
    async_cp16(ldsB + 16, Bm, voffB + 16);
    if (k0 + 32 < K) {
      __builtin_prefetch(A + (size_t)(blkM + rA) * K + k0 + 32 + kbA, 0, 0);
      __builtin_prefetch(Bm + (size_t)(k0 + 32 + rB) * N + blkN + nbB, 0, 0);
    }
    async_wait0();
    __syncthreads();

    v16h af[2], bf[2];
#pragma unroll
    for (int mt = 0; mt < 2; ++mt) {
      const h16* ap = As + (wm + mt * 16 + ln16) * 40;
#pragma unroll
      for (int e = 0; e < 16; ++e) af[mt][e] = ap[frag_k(e, hs)];
    }
#pragma unroll
    for (int nt = 0; nt < 2; ++nt) {
      const h16* bp = Bs + (wn + nt * 16 + ln16);
#pragma unroll
      for (int e = 0; e < 16; ++e) bf[nt][e] = bp[frag_k(e, hs) * 72];
    }
#pragma unroll
    for (int mt = 0; mt < 2; ++mt)
#pragma unroll
      for (int nt = 0; nt < 2; ++nt) acc[mt][nt] = wmma_f16(af[mt], bf[nt], acc[mt][nt]);
    __syncthreads();
  }

#pragma unroll
  for (int mt = 0; mt < 2; ++mt) {
#pragma unroll
    for (int nt = 0; nt < 2; ++nt) {
      const int col = blkN + wn + nt * 16 + ln16;
      float s = 0.f, sh = 0.f, bsum = 0.f;
      if (EPI == 0) {
        s = bn_g[col] * rsqrtf(bn_v[col] + 1e-5f);
        sh = bn_b[col] - bn_m[col] * s;
      } else {
        bsum = bias0 ? bias0[col] : 0.f;
        if (bias1) bsum += bias1[col];
      }
#pragma unroll
      for (int j = 0; j < 8; ++j) {
        const int row = blkM + wm + mt * 16 + j + 8 * hs;
        float v = acc[mt][nt][j];
        if (EPI == 0) {
          v = v * s + sh;
          v = 0.5f * v * (1.0f + erff(v * 0.70710678118f));  // exact GELU
          Dh[(size_t)row * ldD + colOff + col] = (h16)v;
        } else if (EPI == 1) {
          const int head = col >> 6, dd = col & 63;
          const int bb = row / 576, tk = row % 576;
          Dh[(((size_t)bb * 12 + head) * 576 + tk) * 64 + dd] = (h16)(v + bsum);
        } else {
          Df[(size_t)row * ldD + col] = v + bsum;
        }
      }
    }
  }
}

// ---------------------------------------------------------------------------
// Flash attention per (batch*head, 64-query-tile). Q/K/V in (B*12, 576, 64) f16.
// Out written to (B, 576, 768) fp16 token-major for the final projection GEMM.
// ---------------------------------------------------------------------------
__launch_bounds__(128)
__global__ void attn_kernel(const h16* __restrict__ Q, const h16* __restrict__ Km,
                            const h16* __restrict__ V, h16* __restrict__ Out) {
  __shared__ h16 Qs[64 * 72];
  __shared__ h16 Ks[64 * 72];
  __shared__ h16 Vs[64 * 72];
  __shared__ h16 Ps[4 * 16 * 64];  // per-wave P scratch

  const int tid = threadIdx.x;
  const int wv = tid >> 5, lane = tid & 31;
  const int hs = lane >> 4, ln16 = lane & 15;
  const int bh = blockIdx.x / 9;
  const int qt = blockIdx.x % 9;

  const h16* Qb = Q + (size_t)bh * 576 * 64;
  const h16* Kb = Km + (size_t)bh * 576 * 64;
  const h16* Vb = V + (size_t)bh * 576 * 64;

  const int r = tid >> 1, cb = (tid & 1) * 32;  // 64 rows x 64 halfs, 32 halfs/thread
  {  // stage Q tile (64x64) async
    const uint32_t voff = (uint32_t)((((size_t)(qt * 64 + r)) * 64 + cb) * sizeof(h16));
    const uint32_t ldsq = lds_off(Qs + r * 72 + cb);
#pragma unroll
    for (int q8 = 0; q8 < 4; ++q8)
      async_cp16(ldsq + 16 * q8, Qb, voff + 16 * q8);
  }
  async_wait0();
  __syncthreads();

  v16h aq0, aq1;  // Q A-frags for k=0..31 / 32..63
  {
    const h16* qp = Qs + (wv * 16 + ln16) * 72;
#pragma unroll
    for (int e = 0; e < 16; ++e) {
      aq0[e] = qp[frag_k(e, hs)];
      aq1[e] = qp[32 + frag_k(e, hs)];
    }
  }

  v8f vz = {};
  float rmax[8], rsum[8];
  v8f Oacc[4];
#pragma unroll
  for (int j = 0; j < 8; ++j) { rmax[j] = -1e30f; rsum[j] = 0.f; }
#pragma unroll
  for (int nt = 0; nt < 4; ++nt) Oacc[nt] = vz;

  const uint32_t ldsk = lds_off(Ks + r * 72 + cb);
  const uint32_t ldsv = lds_off(Vs + r * 72 + cb);

  for (int kt = 0; kt < 9; ++kt) {
    {  // stage K,V tiles async
      const uint32_t voff = (uint32_t)((((size_t)(kt * 64 + r)) * 64 + cb) * sizeof(h16));
#pragma unroll
      for (int q8 = 0; q8 < 4; ++q8) {
        async_cp16(ldsk + 16 * q8, Kb, voff + 16 * q8);
        async_cp16(ldsv + 16 * q8, Vb, voff + 16 * q8);
      }
    }
    async_wait0();
    __syncthreads();

    // S = Q K^T for 16 query rows x 64 keys (B = K^T: k-index = d, n = key)
    v8f S[4];
#pragma unroll
    for (int nt = 0; nt < 4; ++nt) {
      v16h bk0, bk1;
      const h16* kp = Ks + (nt * 16 + ln16) * 72;
#pragma unroll
      for (int e = 0; e < 16; ++e) {
        bk0[e] = kp[frag_k(e, hs)];
        bk1[e] = kp[32 + frag_k(e, hs)];
      }
      v8f s = vz;
      s = wmma_f16(aq0, bk0, s);
      s = wmma_f16(aq1, bk1, s);
      S[nt] = s;
    }
#pragma unroll
    for (int nt = 0; nt < 4; ++nt)
#pragma unroll
      for (int j = 0; j < 8; ++j) S[nt][j] *= 0.125f;  // 1/sqrt(64)

    // online softmax; row j lives in 16-lane group -> xor-reduce masks 1..8
    float mnew[8];
#pragma unroll
    for (int j = 0; j < 8; ++j) {
      float mx = fmaxf(fmaxf(S[0][j], S[1][j]), fmaxf(S[2][j], S[3][j]));
#pragma unroll
      for (int d = 1; d < 16; d <<= 1) mx = fmaxf(mx, __shfl_xor(mx, d, 32));
      mnew[j] = fmaxf(rmax[j], mx);
    }
    float psum[8];
#pragma unroll
    for (int j = 0; j < 8; ++j) {
      const float al = __expf(rmax[j] - mnew[j]);
      rsum[j] *= al;
      psum[j] = 0.f;
#pragma unroll
      for (int nt = 0; nt < 4; ++nt) {
        Oacc[nt][j] *= al;
        const float p = __expf(S[nt][j] - mnew[j]);
        S[nt][j] = p;
        psum[j] += p;
      }
    }
#pragma unroll
    for (int j = 0; j < 8; ++j) {
      float ps = psum[j];
#pragma unroll
      for (int d = 1; d < 16; d <<= 1) ps += __shfl_xor(ps, d, 32);
      rsum[j] += ps;
      rmax[j] = mnew[j];
    }

    // stage P (C-layout -> row-major LDS), wave-private region
    h16* pwl = Ps + wv * 16 * 64;
#pragma unroll
    for (int nt = 0; nt < 4; ++nt)
#pragma unroll
      for (int j = 0; j < 8; ++j)
        pwl[(j + 8 * hs) * 64 + nt * 16 + ln16] = (h16)S[nt][j];
    asm volatile("s_wait_dscnt 0x0" ::: "memory");  // same-wave LDS RAW

    v16h pa0, pa1;  // P as A operand
    {
      const h16* pp = Ps + wv * 16 * 64 + ln16 * 64;
#pragma unroll
      for (int e = 0; e < 16; ++e) {
        pa0[e] = pp[frag_k(e, hs)];
        pa1[e] = pp[32 + frag_k(e, hs)];
      }
    }
#pragma unroll
    for (int nt = 0; nt < 4; ++nt) {  // O += P * V  (B: k = key, n = d)
      v16h bv0, bv1;
      const h16* vp = Vs + nt * 16 + ln16;
#pragma unroll
      for (int e = 0; e < 16; ++e) {
        bv0[e] = vp[frag_k(e, hs) * 72];
        bv1[e] = vp[(32 + frag_k(e, hs)) * 72];
      }
      Oacc[nt] = wmma_f16(pa0, bv0, Oacc[nt]);
      Oacc[nt] = wmma_f16(pa1, bv1, Oacc[nt]);
    }
    __syncthreads();
  }

  const int b = bh / 12, head = bh % 12;
#pragma unroll
  for (int nt = 0; nt < 4; ++nt)
#pragma unroll
    for (int j = 0; j < 8; ++j) {
      const int row = qt * 64 + wv * 16 + j + 8 * hs;
      const int col = head * 64 + nt * 16 + ln16;
      Out[((size_t)b * 576 + row) * 768 + col] = (h16)(Oacc[nt][j] / rsum[j]);
    }
}

// ---------------------------------------------------------------------------
// Host orchestration
// d_in: 0=x 1=H 2=W, 3..22 q_params{dw,pw,g,b,m,v}x3+{proj_w,proj_b},
//       23..42 k_params, 43..62 v_params, 63=q_bias 64=v_bias 65=proj_w 66=proj_b
// ---------------------------------------------------------------------------
extern "C" void kernel_launch(void* const* d_in, const int* in_sizes, int n_in,
                              void* d_out, int out_size, void* d_ws, size_t ws_size,
                              hipStream_t stream) {
  const int B = 16, C = 768, TOK = B * 576;  // 9216 tokens

  const float* x = (const float*)d_in[0];
  char* wsb = (char*)d_ws;
  size_t off = 0;
  auto salloc = [&](size_t bytes) -> void* {
    void* p = wsb + off;
    off = (off + bytes + 255) & ~(size_t)255;
    return p;
  };
  h16* ybuf = (h16*)salloc((size_t)TOK * C * 2);  // depthwise output (reused)
  h16* cat  = (h16*)salloc((size_t)TOK * C * 2);  // branch concat (reused per proj)
  h16* qh   = (h16*)salloc((size_t)TOK * C * 2);  // (B,head,576,64)
  h16* kh   = (h16*)salloc((size_t)TOK * C * 2);
  h16* vh   = (h16*)salloc((size_t)TOK * C * 2);
  h16* aout = (h16*)salloc((size_t)TOK * C * 2);  // attention out, token-major
  h16* Bh   = (h16*)salloc((size_t)C * C * 2);    // fp16 weight staging (reused)

  const int KSz[3] = {1, 3, 5};
  h16* qkv[3] = {qh, kh, vh};
  const float* qb = (const float*)d_in[63];
  const float* vb = (const float*)d_in[64];

  for (int p = 0; p < 3; ++p) {
    const int base = 3 + 20 * p;
    for (int i = 0; i < 3; ++i) {
      const float* dw  = (const float*)d_in[base + 6 * i + 0];
      const float* pwt = (const float*)d_in[base + 6 * i + 1];
      const float* g   = (const float*)d_in[base + 6 * i + 2];
      const float* bb  = (const float*)d_in[base + 6 * i + 3];
      const float* mm  = (const float*)d_in[base + 6 * i + 4];
      const float* vv  = (const float*)d_in[base + 6 * i + 5];

      const int total = TOK * C;
      dwconv_kernel<<<(total + 255) / 256, 256, 0, stream>>>(x, dw, ybuf, KSz[i], total);
      const int wtot = C * 256;
      wcvt_kernel<<<(wtot + 255) / 256, 256, 0, stream>>>(pwt, Bh, 256, C, wtot);
      dim3 g1(TOK / 64, 256 / 64);
      gemm_wmma_kernel<0><<<g1, 128, 0, stream>>>(ybuf, Bh, TOK, 256, C,
                                                  cat, nullptr, C, i * 256,
                                                  g, bb, mm, vv, nullptr, nullptr);
    }
    const float* pw = (const float*)d_in[base + 18];
    const float* pb = (const float*)d_in[base + 19];
    const int wtot = C * C;
    wcvt_kernel<<<(wtot + 255) / 256, 256, 0, stream>>>(pw, Bh, C, C, wtot);
    const float* b1 = (p == 0) ? qb : (p == 2) ? vb : nullptr;
    dim3 g2(TOK / 64, C / 64);
    gemm_wmma_kernel<1><<<g2, 128, 0, stream>>>(cat, Bh, TOK, C, C,
                                                qkv[p], nullptr, C, 0,
                                                nullptr, nullptr, nullptr, nullptr, pb, b1);
  }

  attn_kernel<<<B * 12 * 9, 128, 0, stream>>>(qh, kh, vh, aout);

  const float* opw = (const float*)d_in[65];
  const float* opb = (const float*)d_in[66];
  const int wtot = C * C;
  wcvt_kernel<<<(wtot + 255) / 256, 256, 0, stream>>>(opw, Bh, C, C, wtot);
  dim3 g3(TOK / 64, C / 64);
  gemm_wmma_kernel<2><<<g3, 128, 0, stream>>>(aout, Bh, TOK, C, C,
                                              nullptr, (float*)d_out, C, 0,
                                              nullptr, nullptr, nullptr, nullptr, opb, nullptr);
  (void)in_sizes; (void)n_in; (void)out_size; (void)ws_size;
}
